// Net_76570676953127
// MI455X (gfx1250) — matrix operand order repaired and checked by reading, not compiled, not run on hardware
//
#include <hip/hip_runtime.h>
#include <hip/hip_bf16.h>
#include <math.h>

// ---------------- problem dims ----------------
#define BB      8
#define SS      512
#define DM      512
#define NLAYERS 4
#define ED      1024         // EXPAND * D_MODEL
#define DSTATE  16
#define DCONV   4
#define DTRANK  32
#define MTOK    (BB * SS)    // 4096 token rows

typedef __attribute__((ext_vector_type(16))) __bf16 v16bf;
typedef __attribute__((ext_vector_type(8)))  float  v8f;

// round-to-nearest-even f32 -> bf16 (stored as ushort)
__device__ __forceinline__ unsigned short f2bf(float f) {
    unsigned int u = __float_as_uint(f);
    unsigned int r = u + 0x7FFFu + ((u >> 16) & 1u);
    return (unsigned short)(r >> 16);
}

// ---------------- workspace layout (bytes) ----------------
#define OFF_XCUR  ((size_t)0)                                   // f32 [MTOK, DM]
#define OFF_XN16  (OFF_XCUR + (size_t)MTOK * DM * 4)            // bf16 [MTOK, DM]
#define OFF_XZ    (OFF_XN16 + (size_t)MTOK * DM * 2)            // f32 [MTOK, 2*ED]
#define OFF_XC    (OFF_XZ   + (size_t)MTOK * 2 * ED * 4)        // f32 [MTOK, ED]
#define OFF_XC16  (OFF_XC   + (size_t)MTOK * ED * 4)            // bf16 [MTOK, ED]
#define OFF_DBC   (OFF_XC16 + (size_t)MTOK * ED * 2)            // f32 [MTOK, 64]
#define OFF_DT16  (OFF_DBC  + (size_t)MTOK * 64 * 4)            // bf16 [MTOK, 32]
#define OFF_DELTA (OFF_DT16 + (size_t)MTOK * DTRANK * 2)        // f32 [MTOK, ED]
#define OFF_Y16   (OFF_DELTA+ (size_t)MTOK * ED * 4)            // bf16 [MTOK, ED]
#define OFF_WTIN  (OFF_Y16  + (size_t)MTOK * ED * 2)            // bf16 [2*ED, DM]
#define OFF_WTXP  (OFF_WTIN + (size_t)DM * 2 * ED * 2)          // bf16 [64, ED]
#define OFF_WTDT  (OFF_WTXP + (size_t)ED * 64 * 2)              // bf16 [ED, 32]
#define OFF_WTOUT (OFF_WTDT + (size_t)DTRANK * ED * 2)          // bf16 [DM, ED]

// ---------------- convert, no transpose: W[N,K] f32 -> Wt[N,K] bf16 ----------------
__global__ void k_transpose_bf16(const float* __restrict__ W,
                                 unsigned short* __restrict__ Wt,
                                 int N, int K) {
    int idx = blockIdx.x * 256 + threadIdx.x;
    if (idx >= N * K) return;
    int n = idx / K, k = idx % K;
    Wt[(size_t)n * K + k] = f2bf(W[idx]);
}

// ---------------- RMSNorm over DM, output bf16 ----------------
__global__ void k_rmsnorm(const float* __restrict__ x,
                          const float* __restrict__ w,
                          unsigned short* __restrict__ out16) {
    __shared__ float red[256];
    int tok = blockIdx.x;                    // [0, MTOK)
    int t = threadIdx.x;                     // 256 threads, 2 elems each
    const float* row = x + (size_t)tok * DM;
    float a = row[t], b = row[t + 256];
    red[t] = a * a + b * b;
    __syncthreads();
    for (int s = 128; s > 0; s >>= 1) {
        if (t < s) red[t] += red[t + s];
        __syncthreads();
    }
    float rms = rsqrtf(red[0] * (1.0f / DM) + 1e-5f);
    out16[(size_t)tok * DM + t]       = f2bf(a * rms * w[t]);
    out16[(size_t)tok * DM + t + 256] = f2bf(b * rms * w[t + 256]);
}

// ---------------- bf16 WMMA GEMM:  C[M,N] = A[M,K] (bf16) * Wt[N,K]^T (bf16) ----------------
// Register-blocked: each wave computes a 32(M) x 64(N) macro-tile
//   = 2 A-fragments x 4 B-fragments -> 8 v_wmma per 32-K step.
// __launch_bounds__(256, 1): allow full VGPR budget (no spills); latency hiding
// comes from the 12-load clause ahead of the 8 WMMAs and the 8 waves per block.
// #pragma unroll 1 on the K loop keeps exactly one fragment set live.
// mode 0: C = acc           (plain f32 store)
// mode 1: C = softplus(acc + bias[n])
// mode 2: C = acc + resid[row,n]       (residual; C aliases resid elementwise)
__global__ void __launch_bounds__(256, 1)
k_gemm_wmma(const unsigned short* __restrict__ A,
            const unsigned short* __restrict__ Wt,
            const float* __restrict__ bias,
            const float* __restrict__ resid,
            float* __restrict__ C,
            int M, int N, int K, int mode) {
    int wave = (blockIdx.x * blockDim.x + threadIdx.x) >> 5;  // wave32
    int lane = threadIdx.x & 31;
    int tn_cnt = N >> 6;                                      // N / 64
    int total = (M >> 5) * tn_cnt;                            // (M/32) * (N/64)
    if (wave >= total) return;                                // wave-uniform exit (EXEC all-1 for WMMA)
    int tm = wave / tn_cnt;
    int tn = wave - tm * tn_cnt;

    int half = lane >> 4;                                     // A-frag K-group select
    int mlo = tm * 32 + (lane & 15);
    const unsigned short* arow0 = A + (size_t)mlo * K + half * 8;   // rows tm*32 + 0..15
    const unsigned short* arow1 = arow0 + (size_t)16 * K;           // rows tm*32 + 16..31
    const unsigned short* bbase = Wt + (size_t)(tn * 64 + (lane & 15)) * K + half * 8;  // B frag: lane&15 = N index

    v8f acc[2][4] = {};
    #pragma unroll 1
    for (int kk = 0; kk < K; kk += 32) {
        union { uint4 q[2]; v16bf v; } ua[2], ub[4];
        ua[0].q[0] = *(const uint4*)(arow0 + kk);          // K = kk + 8*half + 0..7
        ua[0].q[1] = *(const uint4*)(arow0 + kk + 16);     // K = kk + 8*half + 16..23
        ua[1].q[0] = *(const uint4*)(arow1 + kk);
        ua[1].q[1] = *(const uint4*)(arow1 + kk + 16);
        const unsigned short* brow = bbase + kk;
        #pragma unroll
        for (int j = 0; j < 4; ++j) {
            ub[j].q[0] = *(const uint4*)(brow + (size_t)j * 16 * K);       // K = kk + 8*half + 0..7
            ub[j].q[1] = *(const uint4*)(brow + (size_t)j * 16 * K + 16);  // K = kk + 8*half + 16..23
        }
        __builtin_prefetch((const void*)(brow + (size_t)32 * N), 0, 3);
        #pragma unroll
        for (int i = 0; i < 2; ++i)
            #pragma unroll
            for (int j = 0; j < 4; ++j)
                acc[i][j] = __builtin_amdgcn_wmma_f32_16x16x32_bf16(
                    false, ua[i].v, false, ub[j].v, (short)0, acc[i][j], false, false);
    }

    // C/D layout: element r of acc -> row = 16*i + 8*half + r, col = 16*j + (lane&15)
    #pragma unroll
    for (int i = 0; i < 2; ++i) {
        int row0 = tm * 32 + i * 16 + half * 8;
        #pragma unroll
        for (int j = 0; j < 4; ++j) {
            int n = tn * 64 + j * 16 + (lane & 15);
            #pragma unroll
            for (int r = 0; r < 8; ++r) {
                float v = acc[i][j][r];
                size_t o = (size_t)(row0 + r) * N + n;
                if (mode == 1) {
                    v += bias[n];
                    v = (v > 20.0f) ? v : log1pf(expf(v));     // softplus
                } else if (mode == 2) {
                    v += resid[o];
                }
                C[o] = v;
            }
        }
    }
}

// ---------------- causal depthwise conv (D_CONV=4) + bias + SiLU ----------------
__global__ void k_conv_silu(const float* __restrict__ xz,       // [MTOK, 2*ED], xb = [:,0:ED]
                            const float* __restrict__ cw,       // [ED, 4] (layer slice)
                            const float* __restrict__ cb,       // [ED]
                            float* __restrict__ xc,
                            unsigned short* __restrict__ xc16) {
    int idx = blockIdx.x * 256 + threadIdx.x;                   // [0, MTOK*ED)
    if (idx >= MTOK * ED) return;
    int e = idx & (ED - 1);
    int tok = idx >> 10;                                        // ED == 1024
    int s = tok & (SS - 1);
    float acc = cb[e];
    #pragma unroll
    for (int t = 0; t < DCONV; ++t) {
        int sp = s - (DCONV - 1) + t;
        if (sp >= 0)
            acc += xz[((size_t)(tok - (s - sp))) * (2 * ED) + e] * cw[e * DCONV + t];
    }
    float v = acc / (1.0f + expf(-acc));                        // silu
    xc[idx] = v;
    xc16[idx] = f2bf(v);
}

// ---------------- slice dt = dbc[:, :32] -> bf16 ----------------
__global__ void k_dt_bf16(const float* __restrict__ dbc, unsigned short* __restrict__ dt16) {
    int idx = blockIdx.x * 256 + threadIdx.x;                   // [0, MTOK*32)
    if (idx >= MTOK * DTRANK) return;
    int row = idx >> 5, c = idx & 31;
    dt16[idx] = f2bf(dbc[(size_t)row * 64 + c]);
}

// ---------------- selective scan, fused with SiLU(z) gate, bf16 output ----------------
// one thread per (b, e); h[16] in registers; per-step Bm/Cm broadcast via LDS
__global__ void k_scan(const float* __restrict__ dbc,           // [MTOK, 64]: B=32..47 C=48..63
                       const float* __restrict__ delta,         // [MTOK, ED]
                       const float* __restrict__ xc,            // [MTOK, ED]
                       const float* __restrict__ xz,            // [MTOK, 2*ED], z = [:,ED:]
                       const float* __restrict__ A_log,         // [ED, 16] (layer slice)
                       const float* __restrict__ Dp,            // [ED]     (layer slice)
                       unsigned short* __restrict__ y16) {
    int b = blockIdx.x >> 2;                                    // 4 blocks per batch (ED/256)
    int e = ((blockIdx.x & 3) << 8) + threadIdx.x;
    float Arow[DSTATE], h[DSTATE];
    #pragma unroll
    for (int n = 0; n < DSTATE; ++n) {
        Arow[n] = -expf(A_log[(size_t)e * DSTATE + n]);
        h[n] = 0.0f;
    }
    float dpe = Dp[e];
    __shared__ float sB[DSTATE], sC[DSTATE];
    for (int s = 0; s < SS; ++s) {
        size_t tok = (size_t)b * SS + s;
        if (threadIdx.x < 32) {
            float v = dbc[tok * 64 + 32 + threadIdx.x];
            if (threadIdx.x < 16) sB[threadIdx.x] = v;
            else                  sC[threadIdx.x - 16] = v;
        }
        __syncthreads();
        float dv = delta[tok * ED + e];
        float xv = xc[tok * ED + e];
        float accy = 0.0f;
        #pragma unroll
        for (int n = 0; n < DSTATE; ++n) {
            float da = expf(dv * Arow[n]);
            h[n] = da * h[n] + dv * sB[n] * xv;
            accy += h[n] * sC[n];
        }
        float z = xz[tok * (2 * ED) + ED + e];
        float gate = z / (1.0f + expf(-z));                     // silu(z)
        y16[tok * ED + e] = f2bf((accy + dpe * xv) * gate);
        __syncthreads();
    }
}

// ---------------- final head: relu(last @ w1^T + b1) @ w2^T + b2 ----------------
__global__ void k_head(const float* __restrict__ xcur,
                       const float* __restrict__ w1, const float* __restrict__ b1,
                       const float* __restrict__ w2, const float* __restrict__ b2,
                       float* __restrict__ out) {
    __shared__ float h1[64];
    int b = blockIdx.x, j = threadIdx.x;                        // 64 threads
    const float* last = xcur + ((size_t)b * SS + (SS - 1)) * DM;
    float s = b1[j];
    for (int i = 0; i < DM; ++i) s += last[i] * w1[j * DM + i];
    h1[j] = fmaxf(s, 0.0f);
    __syncthreads();
    if (j == 0) {
        float o = b2[0];
        for (int k = 0; k < 64; ++k) o += h1[k] * w2[k];
        out[b] = o;
    }
}

// ---------------- launcher ----------------
extern "C" void kernel_launch(void* const* d_in, const int* in_sizes, int n_in,
                              void* d_out, int out_size, void* d_ws, size_t ws_size,
                              hipStream_t stream) {
    const float* x        = (const float*)d_in[0];
    const float* norm_w   = (const float*)d_in[1];
    const float* in_proj  = (const float*)d_in[2];   // [L, 2*ED, DM]
    const float* conv_w   = (const float*)d_in[3];   // [L, ED, 1, 4]
    const float* conv_b   = (const float*)d_in[4];   // [L, ED]
    const float* x_proj   = (const float*)d_in[5];   // [L, 64, ED]
    const float* dt_proj  = (const float*)d_in[6];   // [L, ED, 32]
    const float* dt_b     = (const float*)d_in[7];   // [L, ED]
    const float* A_log    = (const float*)d_in[8];   // [L, ED, 16]
    const float* Dp       = (const float*)d_in[9];   // [L, ED]
    const float* out_proj = (const float*)d_in[10];  // [L, DM, ED]
    const float* out_w1   = (const float*)d_in[11];
    const float* out_b1   = (const float*)d_in[12];
    const float* out_w2   = (const float*)d_in[13];
    const float* out_b2   = (const float*)d_in[14];

    char* ws = (char*)d_ws;
    float*          XCUR  = (float*)(ws + OFF_XCUR);
    unsigned short* XN16  = (unsigned short*)(ws + OFF_XN16);
    float*          XZ    = (float*)(ws + OFF_XZ);
    float*          XC    = (float*)(ws + OFF_XC);
    unsigned short* XC16  = (unsigned short*)(ws + OFF_XC16);
    float*          DBC   = (float*)(ws + OFF_DBC);
    unsigned short* DT16  = (unsigned short*)(ws + OFF_DT16);
    float*          DELTA = (float*)(ws + OFF_DELTA);
    unsigned short* Y16   = (unsigned short*)(ws + OFF_Y16);
    unsigned short* WTIN  = (unsigned short*)(ws + OFF_WTIN);
    unsigned short* WTXP  = (unsigned short*)(ws + OFF_WTXP);
    unsigned short* WTDT  = (unsigned short*)(ws + OFF_WTDT);
    unsigned short* WTOUT = (unsigned short*)(ws + OFF_WTOUT);

    // residual stream starts as the input
    hipMemcpyAsync(XCUR, x, (size_t)MTOK * DM * sizeof(float),
                   hipMemcpyDeviceToDevice, stream);

    for (int l = 0; l < NLAYERS; ++l) {
        // -- per-layer weight f32->bf16 convert --
        k_transpose_bf16<<<(2 * ED * DM + 255) / 256, 256, 0, stream>>>(
            in_proj + (size_t)l * 2 * ED * DM, WTIN, 2 * ED, DM);
        k_transpose_bf16<<<(64 * ED + 255) / 256, 256, 0, stream>>>(
            x_proj + (size_t)l * 64 * ED, WTXP, 64, ED);
        k_transpose_bf16<<<(ED * DTRANK + 255) / 256, 256, 0, stream>>>(
            dt_proj + (size_t)l * ED * DTRANK, WTDT, ED, DTRANK);
        k_transpose_bf16<<<(DM * ED + 255) / 256, 256, 0, stream>>>(
            out_proj + (size_t)l * DM * ED, WTOUT, DM, ED);

        // -- RMSNorm -> bf16 --
        k_rmsnorm<<<MTOK, 256, 0, stream>>>(XCUR, norm_w + (size_t)l * DM, XN16);

        // -- in_proj GEMM: [4096 x 2048] = XN16[4096 x 512] * WTIN --
        {
            int waves = (MTOK / 32) * ((2 * ED) / 64);          // 4096
            k_gemm_wmma<<<waves / 8, 256, 0, stream>>>(
                XN16, WTIN, nullptr, nullptr, XZ, MTOK, 2 * ED, DM, 0);
        }

        // -- causal dwconv + SiLU --
        k_conv_silu<<<(MTOK * ED + 255) / 256, 256, 0, stream>>>(
            XZ, conv_w + (size_t)l * ED * DCONV, conv_b + (size_t)l * ED, XC, XC16);

        // -- x_proj GEMM: [4096 x 64] = XC16[4096 x 1024] * WTXP --
        {
            int waves = (MTOK / 32) * (64 / 64);                // 128
            k_gemm_wmma<<<waves / 8, 256, 0, stream>>>(
                XC16, WTXP, nullptr, nullptr, DBC, MTOK, 64, ED, 0);
        }

        // -- dt slice -> bf16 --
        k_dt_bf16<<<(MTOK * DTRANK + 255) / 256, 256, 0, stream>>>(DBC, DT16);

        // -- dt_proj GEMM + bias + softplus: delta[4096 x 1024], K=32 --
        {
            int waves = (MTOK / 32) * (ED / 64);                // 2048
            k_gemm_wmma<<<waves / 8, 256, 0, stream>>>(
                DT16, WTDT, dt_b + (size_t)l * ED, nullptr, DELTA, MTOK, ED, DTRANK, 1);
        }

        // -- selective scan fused with SiLU(z) gate -> bf16 --
        k_scan<<<BB * (ED / 256), 256, 0, stream>>>(
            DBC, DELTA, XC, XZ,
            A_log + (size_t)l * ED * DSTATE, Dp + (size_t)l * ED, Y16);

        // -- out_proj GEMM + residual: XCUR = Y16 * WTOUT + XCUR --
        {
            int waves = (MTOK / 32) * (DM / 64);                // 1024
            k_gemm_wmma<<<waves / 8, 256, 0, stream>>>(
                Y16, WTOUT, nullptr, XCUR, XCUR, MTOK, DM, ED, 2);
        }
    }

    // -- head --
    k_head<<<BB, 64, 0, stream>>>(XCUR, out_w1, out_b1, out_w2, out_b2, (float*)d_out);
}
